// LSTMModel_4389456576600
// MI455X (gfx1250) — matrix operand order repaired and checked by reading, not compiled, not run on hardware
//
#include <hip/hip_runtime.h>
#include <hip/hip_bf16.h>

typedef __bf16 bf16_t;
typedef __attribute__((ext_vector_type(8)))  __bf16 v8bf;
typedef __attribute__((ext_vector_type(16))) __bf16 v16bf;
typedef __attribute__((ext_vector_type(8)))  float  v8f;

#define B_   64
#define T_   512
#define IN_  256
#define H_   1024
#define G4_  4096
#define KTOT 1280            // IN_ + H_ rows of transposed weights
#define NWG  64              // one workgroup per 16 hidden units
#define KW   (KTOT + 8)      // LDS row stride (halves) for the [W_ih;W_hh] slice
                             // 2576 B = 644 dwords; 644 % 64 == 4 -> conflict-free
#define HSW  (H_ + 8)        // LDS row stride (halves) for the h stage: 2064 B
                             // 516 dwords % 64 == 4 -> 16 rows on disjoint banks

// Build a v16bf WMMA fragment from two contiguous 16-byte chunks.
#define LOAD16(p0, p1)                                                     \
  __builtin_shufflevector(*(const v8bf*)(p0), *(const v8bf*)(p1),          \
                          0,1,2,3,4,5,6,7,8,9,10,11,12,13,14,15)

__device__ __forceinline__ float fsig(float x) {
  return 1.0f / (1.0f + __expf(-x));
}
__device__ __forceinline__ float ftanh(float x) {
  float e = __expf(-2.0f * x);
  return (1.0f - e) / (1.0f + e);
}

// ---- pack W_ih (256x4096) and W_hh (1024x4096) into transposed bf16 Wt[col][k]
__global__ void k_pack_w(const float* __restrict__ Wih,
                         const float* __restrict__ Whh,
                         bf16_t* __restrict__ Wt) {
  int col = blockIdx.x * 256 + threadIdx.x;   // 0..4095 (coalesced reads)
  int k   = blockIdx.y;                       // 0..1279
  float v = (k < IN_) ? Wih[(size_t)k * G4_ + col]
                      : Whh[(size_t)(k - IN_) * G4_ + col];
  Wt[(size_t)col * KTOT + k] = (bf16_t)v;
}

// ---- x fp32 -> bf16 (same [B][T][IN] layout)
__global__ void k_pack_x(const float* __restrict__ x, bf16_t* __restrict__ xb, int n) {
  int i = blockIdx.x * 256 + threadIdx.x;
  if (i < n) xb[i] = (bf16_t)x[i];
}

// ---- zero both h ping-pong buffers (contiguous) + the grid-barrier counter
__global__ void k_zero(bf16_t* __restrict__ h01, unsigned* __restrict__ bar) {
  int i = blockIdx.x * 256 + threadIdx.x;
  if (i < 2 * B_ * H_) h01[i] = (bf16_t)0.0f;
  if (i == 0) *bar = 0u;
}

// ---- persistent fused LSTM scan: 64 WGs x 128 threads (4 waves)
// WG wg owns hidden units j0..j0+15 for all 4 gates (64 columns of [W_ih;W_hh],
// resident in LDS for the whole sequence). wave w owns batch rows 16w..16w+15
// and keeps its cell-state tile in registers. Per step:
//   (1) async-copy this wave's 16 h rows -> LDS stage (ASYNCcnt),
//   (2) x-part GEMM (A from L2, B from LDS) hides the async latency,
//   (3) s_wait_asynccnt 0, (4) h-part GEMM entirely from LDS,
//   (5) pointwise cell update, scatter h, device-wide barrier.
__global__ __launch_bounds__(128, 1) void k_lstm(
    const bf16_t* __restrict__ xb, const bf16_t* __restrict__ Wt,
    const float* __restrict__ bih, const float* __restrict__ bhh,
    bf16_t* __restrict__ h0, bf16_t* __restrict__ h1,
    unsigned* __restrict__ bar)
{
  __shared__ bf16_t sW[64 * KW];          // [W_ih;W_hh] slice, transposed: sW[c][k]
  __shared__ bf16_t sH[4 * 16 * HSW];     // per-wave h-row stage (4 x 33 KB)

  const int tid  = threadIdx.x;
  const int wg   = blockIdx.x;
  const int j0   = wg * 16;
  const int wave = tid >> 5;
  const int lane = tid & 31;
  const int n    = lane & 15;        // B/C/D column
  const int hi   = lane >> 4;        // half-wave select
  const int kh8  = hi * 8;           // A-frag K sub-offset (ISA 16-bit A 16x32 layout)
  const int kb16 = hi * 16;          // unused: B frags also use kh8 (same k order as A)
  const int b0   = wave * 16;
  const int m    = n;                // A row within tile

  // Stage the full fused weight slice (64 cols x 1280 K) into LDS, once.
  for (int idx = tid; idx < 64 * (KTOT / 8); idx += 128) {
    int c  = idx / (KTOT / 8);       // local col 0..63
    int kc = (idx % (KTOT / 8)) * 8; // K chunk 0..1272
    int g  = c >> 4, cn = c & 15;
    const v8bf* src =
        (const v8bf*)(Wt + (size_t)(g * H_ + j0 + cn) * KTOT + kc);
    *(v8bf*)&sW[c * KW + kc] = *src;
  }
  __syncthreads();

  float bias[4];
  #pragma unroll
  for (int g = 0; g < 4; ++g)
    bias[g] = bih[g * H_ + j0 + n] + bhh[g * H_ + j0 + n];

  v8f cc = {0.f,0.f,0.f,0.f,0.f,0.f,0.f,0.f};   // cell-state tile (f32, persistent)

  bf16_t* sHw = sH + (size_t)wave * 16 * HSW;
  // 32-bit LDS byte address of this wave's stage (generic LDS ptr low bits)
  const unsigned shb  = (unsigned)(uintptr_t)(void*)sHw + (unsigned)(lane * 16);
  const bf16_t* xrow  = xb + ((size_t)(b0 + m) * T_) * IN_;

  for (int t = 0; t < T_; ++t) {
    const bf16_t* hprev = (t & 1) ? h1 : h0;
    bf16_t*       hnext = (t & 1) ? h0 : h1;

    // ---- kick off async copy: 16 h rows (32 KB) -> LDS stage.
    // LDS row stride 2064 B vs global 2048 B: bias VDST by +16 B per row;
    // the 24-bit inst offset is added on both the global and LDS side.
    {
      // previous step's LDS reads must have retired before overwriting stage
      asm volatile("s_wait_dscnt 0x0" ::: "memory");
      uint64_t ga = (uint64_t)(uintptr_t)(hprev + (size_t)b0 * H_) +
                    (uint64_t)(unsigned)(lane * 16);
      #pragma unroll
      for (int r = 0; r < 16; ++r) {
        unsigned ld = shb + (unsigned)(r * 16);
        #pragma unroll
        for (int q = 0; q < 4; ++q) {
          asm volatile("global_load_async_to_lds_b128 %0, %1, off offset:%2"
                       :: "v"(ld), "v"(ga), "n"(r * 2048 + q * 512)
                       : "memory");
        }
      }
    }

    v8f acc[4];
    #pragma unroll
    for (int g = 0; g < 4; ++g) acc[g] = (v8f){0.f,0.f,0.f,0.f,0.f,0.f,0.f,0.f};

    // ---- input projection: K = 0..255 (A = x_t from global, B = LDS);
    // runs while the h async-copy is in flight.
    const bf16_t* xr = xrow + (size_t)t * IN_;
    #pragma unroll
    for (int ks = 0; ks < IN_ / 32; ++ks) {
      const int k0 = ks * 32;
      v16bf av = LOAD16(xr + k0 + kh8, xr + k0 + 16 + kh8);
      #pragma unroll
      for (int g = 0; g < 4; ++g) {
        const bf16_t* wr = &sW[(g * 16 + n) * KW + k0 + kh8];
        v16bf bv = LOAD16(wr, wr + 16);
        acc[g] = __builtin_amdgcn_wmma_f32_16x16x32_bf16(
            false, av, false, bv, (short)0, acc[g], false, false);
      }
    }

    // ---- h stage is needed now
    asm volatile("s_wait_asynccnt 0x0" ::: "memory");

    // ---- recurrent projection: K = 0..1023 (A = h_{t-1} and B both from LDS)
    const bf16_t* hrL = sHw + (size_t)m * HSW;
    #pragma unroll 8
    for (int ks = 0; ks < H_ / 32; ++ks) {
      const int k0 = ks * 32;
      v16bf av = LOAD16(hrL + k0 + kh8, hrL + k0 + 16 + kh8);
      #pragma unroll
      for (int g = 0; g < 4; ++g) {
        const bf16_t* wr = &sW[(g * 16 + n) * KW + IN_ + k0 + kh8];
        v16bf bv = LOAD16(wr, wr + 16);
        acc[g] = __builtin_amdgcn_wmma_f32_16x16x32_bf16(
            false, av, false, bv, (short)0, acc[g], false, false);
      }
    }

    // ---- pointwise LSTM cell update; i/f/g/o share lane/element positions
    #pragma unroll
    for (int e = 0; e < 8; ++e) {
      float iv = fsig (acc[0][e] + bias[0]);
      float fv = fsig (acc[1][e] + bias[1]);
      float gv = ftanh(acc[2][e] + bias[2]);
      float ov = fsig (acc[3][e] + bias[3]);
      float c  = fv * cc[e] + iv * gv;
      cc[e] = c;
      float h  = ov * ftanh(c);
      int mg = b0 + hi * 8 + e;                      // C/D layout: M = hi*8 + vgpr
      hnext[(size_t)mg * H_ + j0 + n] = (bf16_t)h;
    }

    // ---- device-wide barrier (monotonic counter, zeroed per launch)
    __threadfence();
    __syncthreads();
    if (tid == 0) {
      atomicAdd(bar, 1u);
      const unsigned target = (unsigned)(NWG * (t + 1));
      while (atomicAdd(bar, 0u) < target) __builtin_amdgcn_s_sleep(8);
    }
    __syncthreads();
    __threadfence();
  }
}

// ---- out[b][o] = h_last[b] . W_out[:,o] + b_out[o]   (tiny, f32 VALU)
__global__ __launch_bounds__(256) void k_out(
    const bf16_t* __restrict__ hfin, const float* __restrict__ Wout,
    const float* __restrict__ bout, float* __restrict__ out)
{
  __shared__ float sh[H_];
  const int b = blockIdx.x, o = threadIdx.x;
  for (int k = o; k < H_; k += 256) sh[k] = (float)hfin[(size_t)b * H_ + k];
  __syncthreads();
  float acc = bout[o];
  #pragma unroll 8
  for (int k = 0; k < H_; ++k) acc += sh[k] * Wout[(size_t)k * 256 + o];
  out[b * 256 + o] = acc;
}

extern "C" void kernel_launch(void* const* d_in, const int* in_sizes, int n_in,
                              void* d_out, int out_size, void* d_ws, size_t ws_size,
                              hipStream_t stream) {
  (void)in_sizes; (void)n_in; (void)out_size; (void)ws_size;
  const float* x    = (const float*)d_in[0];
  const float* Wih  = (const float*)d_in[1];
  const float* Whh  = (const float*)d_in[2];
  const float* bih  = (const float*)d_in[3];
  const float* bhh  = (const float*)d_in[4];
  const float* Wout = (const float*)d_in[5];
  const float* bout = (const float*)d_in[6];
  float* out = (float*)d_out;

  char* ws = (char*)d_ws;
  size_t off = 0;
  bf16_t* Wt = (bf16_t*)(ws + off); off += (size_t)G4_ * KTOT * sizeof(bf16_t);     // 10 MB
  bf16_t* xb = (bf16_t*)(ws + off); off += (size_t)B_ * T_ * IN_ * sizeof(bf16_t);  // 16 MB
  bf16_t* h0 = (bf16_t*)(ws + off); off += (size_t)B_ * H_ * sizeof(bf16_t);        // 128 KB
  bf16_t* h1 = (bf16_t*)(ws + off); off += (size_t)B_ * H_ * sizeof(bf16_t);        // 128 KB
  unsigned* bar = (unsigned*)(ws + off);

  k_pack_w<<<dim3(G4_ / 256, KTOT), 256, 0, stream>>>(Wih, Whh, Wt);
  const int nx = B_ * T_ * IN_;
  k_pack_x<<<(nx + 255) / 256, 256, 0, stream>>>(x, xb, nx);
  k_zero<<<(2 * B_ * H_ + 255) / 256, 256, 0, stream>>>(h0, bar);  // h0,h1 contiguous
  k_lstm<<<NWG, 128, 0, stream>>>(xb, Wt, bih, bhh, h0, h1, bar);
  k_out<<<B_, 256, 0, stream>>>(h0, Wout, bout, out);               // final h is in h0
}
